// SingleNNModel_34797825032476
// MI455X (gfx1250) — matrix operand order repaired and checked by reading, not compiled, not run on hardware
//
#include <hip/hip_runtime.h>
#include <hip/hip_bf16.h>

// ---------------------------------------------------------------------------
// Fused 4-layer MLP (256 -> 512 -> 512 -> 256 -> 1, SiLU) for gfx1250.
// bf16 WMMA (v_wmma_f32_16x16x32_bf16) with f32 accumulation; activations
// stay in LDS for the whole block tile; weights pre-packed to bf16 fragments.
//
// Per-wave register tile: 2 m-tiles x 4 n-tiles (8 accumulator chains).
// K-loop fully unrolled with all fragment loads using immediate offsets from
// one per-group base address -> no inner-loop VALU -> no WMMA WAR v_nops.
// ---------------------------------------------------------------------------

typedef __attribute__((ext_vector_type(16))) __bf16 v16bf;
typedef __attribute__((ext_vector_type(8)))  __bf16 v8bf;
typedef __attribute__((ext_vector_type(8)))  float  v8f;

#define ROWS_PER_BLOCK 64
#define LDS_STRIDE     520   // 512 + 8 bf16 pad (16B) -> rows start 4 banks apart
#define BLOCK_THREADS  256
#define D_IN           256
#define H1             512
#define H2             512
#define H3             256

#define FRAG_BYTES     1024  // one 32x16 bf16 B-fragment: 32 lanes * 32 B

union AFrag { v16bf v; v8bf h[2]; };

__device__ __forceinline__ float silu_f(float x) {
    return x / (1.0f + __expf(-x));
}

// ---------------------------------------------------------------------------
// Pack a row-major fp32 weight matrix W[K][N] into bf16 WMMA B-fragments:
//   Wp[((nt*(K/32) + kt)*32 + lane)*16 + e] = bf16( W[kt*32 + 8*(lane/16) + e%8 + 16*(e/8)][nt*16 + lane%16] )
// (B-matrix layout: lane%16 indexes N, lane half and the 8 VGPRs /
//  16 bf16 elements index K — per CDNA5 ISA B-matrix striping.)
// ---------------------------------------------------------------------------
__global__ void pack_weights_bf16(const float* __restrict__ W,
                                  __bf16* __restrict__ Wp,
                                  int K, int N) {
    int idx = blockIdx.x * blockDim.x + threadIdx.x;
    if (idx >= K * N) return;
    int e    = idx & 15;
    int lane = (idx >> 4) & 31;
    int frag = idx >> 9;
    int KT   = K >> 5;
    int kt   = frag % KT;
    int nt   = frag / KT;
    int k    = kt * 32 + (e & 7) + 8 * (lane >> 4) + 16 * (e >> 3);
    int n    = nt * 16 + (lane & 15);
    Wp[idx] = (__bf16)W[(size_t)k * N + n];
}

// ---------------------------------------------------------------------------
// One dense layer on a 64-row LDS tile:
//   out[64 x N] = silu( in[64 x K] @ W[K x N] + bias )
// 8 waves = 2 m-pairs x 4 n-quarters. Each wave: 2 m-tiles x (N/16/4) n-tiles,
// processed as groups of 4 n-tiles with a 2x4 accumulator block.
// ---------------------------------------------------------------------------
template<int KT>   // KT = K / 32
__device__ __forceinline__ void mlp_layer(
    const __bf16* __restrict__ Wp, const float* __restrict__ bias,
    const __bf16* bufIn, __bf16* bufOut, int NT /* = N/16 */,
    int wave, int lane)
{
    const int mp   = wave & 1;     // m-pair: m-tiles {2mp, 2mp+1}
    const int nq   = wave >> 1;    // n-quarter 0..3
    const int r    = lane & 15;
    const int half = lane >> 4;
    const __bf16* aRow0 = bufIn + ((2 * mp + 0) * 16 + r) * LDS_STRIDE + 8 * half;
    const __bf16* aRow1 = bufIn + ((2 * mp + 1) * 16 + r) * LDS_STRIDE + 8 * half;
    const char* WpB = (const char*)Wp;
    const int laneOff = lane * 32;              // bytes within a fragment
    const int ntPerWave = NT >> 2;

    for (int g = 0; g < ntPerWave; g += 4) {
        const int ntBase = nq * ntPerWave + g;
        // Single per-group base; every load below is base + immediate offset.
        const char* bBase = WpB + (size_t)ntBase * KT * FRAG_BYTES + laneOff;

        v8f acc[2][4];
        #pragma unroll
        for (int i = 0; i < 2; ++i)
            #pragma unroll
            for (int j = 0; j < 4; ++j)
                acc[i][j] = (v8f){0.f,0.f,0.f,0.f,0.f,0.f,0.f,0.f};

        #pragma unroll
        for (int kt = 0; kt < KT; ++kt) {
            // 16-bit A-matrix 16x32 layout: lane = row (mod 16), half selects
            // K sub-bands: e=0..7 -> K = 8*half + e ; e=8..15 -> K = 16+8*half+(e-8).
            AFrag a0, a1;
            a0.h[0] = *(const v8bf*)(aRow0 + kt * 32);
            a0.h[1] = *(const v8bf*)(aRow0 + kt * 32 + 16);
            a1.h[0] = *(const v8bf*)(aRow1 + kt * 32);
            a1.h[1] = *(const v8bf*)(aRow1 + kt * 32 + 16);
            #pragma unroll
            for (int j = 0; j < 4; ++j) {
                const v16bf b = *(const v16bf*)(bBase + (j * KT + kt) * FRAG_BYTES);
                acc[0][j] = __builtin_amdgcn_wmma_f32_16x16x32_bf16(
                    false, a0.v, false, b, (short)0, acc[0][j], false, false);
                acc[1][j] = __builtin_amdgcn_wmma_f32_16x16x32_bf16(
                    false, a1.v, false, b, (short)0, acc[1][j], false, false);
            }
        }

        // D layout: lanes 0-15 -> n=lane, m=v ; lanes 16-31 -> n=lane-16, m=8+v.
        #pragma unroll
        for (int j = 0; j < 4; ++j) {
            const int ncol = (ntBase + j) * 16 + r;
            const float bv = bias[ncol];
            #pragma unroll
            for (int i = 0; i < 2; ++i) {
                __bf16* dst = bufOut + ((2 * mp + i) * 16 + half * 8) * LDS_STRIDE + ncol;
                #pragma unroll
                for (int v = 0; v < 8; ++v) {
                    float x = silu_f(acc[i][j][v] + bv);
                    dst[v * LDS_STRIDE] = (__bf16)x;
                }
            }
        }
    }
}

// ---------------------------------------------------------------------------
// Fused forward kernel: one block = 64 atoms through all 4 layers.
// ---------------------------------------------------------------------------
__global__ void __launch_bounds__(BLOCK_THREADS)
mlp_forward_kernel(const float* __restrict__ density,
                   const int*   __restrict__ species,
                   const float* __restrict__ b1,
                   const float* __restrict__ b2,
                   const float* __restrict__ b3,
                   const float* __restrict__ W4,
                   const float* __restrict__ b4,
                   const __bf16* __restrict__ W1p,
                   const __bf16* __restrict__ W2p,
                   const __bf16* __restrict__ W3p,
                   float* __restrict__ out,
                   int n_atoms)
{
    extern __shared__ __bf16 smem[];
    __bf16* bufA = smem;                                // 64 x LDS_STRIDE
    __bf16* bufB = smem + ROWS_PER_BLOCK * LDS_STRIDE;  // 64 x LDS_STRIDE

    const int tid  = threadIdx.x;
    const int wave = tid >> 5;
    const int lane = tid & 31;
    const int row0 = blockIdx.x * ROWS_PER_BLOCK;

    // Stage input tile -> LDS as bf16 (coalesced b32 loads, zero-pad tail rows)
    for (int idx = tid; idx < ROWS_PER_BLOCK * D_IN; idx += BLOCK_THREADS) {
        int r = idx >> 8;         // / 256
        int c = idx & (D_IN - 1);
        int g = row0 + r;
        float v = (g < n_atoms) ? density[(size_t)g * D_IN + c] : 0.0f;
        bufA[r * LDS_STRIDE + c] = (__bf16)v;
    }
    __syncthreads();

    mlp_layer<D_IN / 32>(W1p, b1, bufA, bufB, H1 / 16, wave, lane);  // 256 -> 512
    __syncthreads();
    mlp_layer<H1 / 32>(W2p, b2, bufB, bufA, H2 / 16, wave, lane);    // 512 -> 512
    __syncthreads();
    mlp_layer<H2 / 32>(W3p, b3, bufA, bufB, H3 / 16, wave, lane);    // 512 -> 256
    __syncthreads();

    // Layer 4: N=1 -> per-row dot product + bias + species mask + store.
    if (tid < ROWS_PER_BLOCK) {
        int g = row0 + tid;
        if (g < n_atoms) {
            const __bf16* h = bufB + tid * LDS_STRIDE;
            float acc = 0.0f;
            #pragma unroll 8
            for (int k = 0; k < H3; ++k)
                acc += (float)h[k] * W4[k];
            acc += b4[0];
            out[g] = (species[g] >= 0) ? acc : 0.0f;   // species > -0.5
        }
    }
}

// ---------------------------------------------------------------------------
// Host launch. Inputs: 0 density, 1 species, 2 W1, 3 b1, 4 W2, 5 b2,
//                      6 W3, 7 b3, 8 W4, 9 b4.
// d_ws layout: W1p[256KB] | W2p[512KB] | W3p[256KB]  (1 MB total).
// ---------------------------------------------------------------------------
extern "C" void kernel_launch(void* const* d_in, const int* in_sizes, int n_in,
                              void* d_out, int out_size, void* d_ws, size_t ws_size,
                              hipStream_t stream) {
    const float* density = (const float*)d_in[0];
    const int*   species = (const int*)  d_in[1];
    const float* W1 = (const float*)d_in[2];
    const float* b1 = (const float*)d_in[3];
    const float* W2 = (const float*)d_in[4];
    const float* b2 = (const float*)d_in[5];
    const float* W3 = (const float*)d_in[6];
    const float* b3 = (const float*)d_in[7];
    const float* W4 = (const float*)d_in[8];
    const float* b4 = (const float*)d_in[9];

    const int n_atoms = in_sizes[1];   // species count == N

    __bf16* W1p = (__bf16*)d_ws;                              // 256*512 bf16
    __bf16* W2p = (__bf16*)((char*)d_ws + 262144);            // 512*512 bf16
    __bf16* W3p = (__bf16*)((char*)d_ws + 262144 + 524288);   // 512*256 bf16

    pack_weights_bf16<<<(D_IN * H1 + 255) / 256, 256, 0, stream>>>(W1, W1p, D_IN, H1);
    pack_weights_bf16<<<(H1 * H2 + 255) / 256, 256, 0, stream>>>(W2, W2p, H1, H2);
    pack_weights_bf16<<<(H2 * H3 + 255) / 256, 256, 0, stream>>>(W3, W3p, H2, H3);

    const int smem_bytes = 2 * ROWS_PER_BLOCK * LDS_STRIDE * (int)sizeof(__bf16);
    (void)hipFuncSetAttribute((const void*)mlp_forward_kernel,
                              hipFuncAttributeMaxDynamicSharedMemorySize,
                              smem_bytes);

    const int grid = (n_atoms + ROWS_PER_BLOCK - 1) / ROWS_PER_BLOCK;
    mlp_forward_kernel<<<grid, BLOCK_THREADS, smem_bytes, stream>>>(
        density, species, b1, b2, b3, W4, b4, W1p, W2p, W3p,
        (float*)d_out, n_atoms);
}
